// Seq2SeqDecoder_12584254177987
// MI455X (gfx1250) — matrix operand order repaired and checked by reading, not compiled, not run on hardware
//
#include <hip/hip_runtime.h>
#include <hip/hip_bf16.h>
#include <math.h>

// Problem constants (match reference)
#define B_   32
#define T_   2048
#define ENC_ 512
#define U_   64
#define V_   128
#define D_   256
#define K_   128
#define VAL_ 256

typedef __attribute__((ext_vector_type(16))) __bf16   v16bf;
typedef __attribute__((ext_vector_type(8)))  float    v8f;
typedef __attribute__((ext_vector_type(8)))  unsigned v8u;
typedef __attribute__((ext_vector_type(4)))  unsigned v4u;

static __device__ __forceinline__ unsigned f32_to_bf16_bits(float f) {
  unsigned u = __builtin_bit_cast(unsigned, f);
  return (u + 0x7FFFu + ((u >> 16) & 1u)) >> 16;   // round-to-nearest-even
}
static __device__ __forceinline__ unsigned pack2_bf16(float a, float b) {
  return f32_to_bf16_bits(a) | (f32_to_bf16_bits(b) << 16);
}
static __device__ __forceinline__ float sigmoidf_(float x) {
  return 1.0f / (1.0f + __expf(-x));
}

// ---------------------------------------------------------------------------
// Weight pack: src (N x Kd) f32 row-major -> dst (Kd/2 x N) u32; each u32 is a
// bf16 pair (k, k+1) for output column n (transposed, K-pair-major), matching
// the WMMA B-matrix per-lane consumption below.
// ---------------------------------------------------------------------------
__global__ void pack_w_kernel(const float* __restrict__ src,
                              unsigned* __restrict__ dst, int N, int Kd) {
  int idx = blockIdx.x * blockDim.x + threadIdx.x;
  int total = (Kd >> 1) * N;
  if (idx >= total) return;
  int n  = idx % N;
  int kp = idx / N;
  dst[idx] = pack2_bf16(src[n * Kd + 2 * kp], src[n * Kd + 2 * kp + 1]);
}

// ---------------------------------------------------------------------------
// Activation pack: X (M x Kd) f32 row-major -> Apk (M x Kd/2) u32 bf16 pairs.
// One HBM pass; removes all conversion VALU from the big GEMM inner loops.
// ---------------------------------------------------------------------------
__global__ void pack_x_kernel(const float* __restrict__ X,
                              unsigned* __restrict__ Apk, long total_pairs) {
  long idx = (long)blockIdx.x * blockDim.x + threadIdx.x;
  if (idx >= total_pairs) return;
  Apk[idx] = pack2_bf16(X[2 * idx], X[2 * idx + 1]);
}

// ---------------------------------------------------------------------------
// Big WMMA GEMM, pre-packed bf16 A:  C(M x N) = A(M x Kd) @ Wpk + bias.
// One wave per 16x32 C block: two N tiles share one A fragment; K loop
// unrolled 2x -> 4 independent v_wmma issues per iteration.  A fragment is
// two b128 loads; N/Kd are compile-time so all B-row offsets become
// immediate instruction offsets (no per-fragment 64-bit pointer adds).
// grid = (N/32, M/16), block = 32.
// ---------------------------------------------------------------------------
template <int N, int Kd>
__global__ void gemm_wmma_pk(const unsigned* __restrict__ Apk, int ldapk,
                             const unsigned* __restrict__ Wpk,
                             const float* __restrict__ bias,
                             float* __restrict__ C, int ldc) {
  const int lane = threadIdx.x & 31;
  const int half = lane >> 4;
  const int l    = lane & 15;
  const int tm   = blockIdx.y << 4;
  const int tn   = blockIdx.x << 5;
  const unsigned* arow = Apk + (tm + l) * ldapk + (half << 3);
  const unsigned* wcol = Wpk + tn + l + (half << 3) * N;  // kb base = half*16

  v8f acc0 = {}, acc1 = {};
#pragma unroll 2
  for (int k0 = 0; k0 < Kd; k0 += 32) {
    // A layout (16-bit A 16x32): per lane, 8 consecutive dword
    // pairs: k0/2 + 8*half + {0..7}.
    const v4u a0 = *(const v4u*)(arow + (k0 >> 1));
    const v4u a1 = *(const v4u*)(arow + (k0 >> 1) + 4);
    v8u au;
    au[0] = a0[0]; au[1] = a0[1]; au[2] = a0[2]; au[3] = a0[3];
    au[4] = a1[0]; au[5] = a1[1]; au[6] = a1[2]; au[7] = a1[3];

    // B layout (16-bit B 32x16): lane = column n, v holds K=2v,2v+1 (+16 upper)
    v8u bu0, bu1;
    const unsigned* wk = wcol + (k0 >> 1) * N;
#pragma unroll
    for (int v = 0; v < 8; ++v) {
      bu0[v] = wk[v * N];         // compile-time immediate offsets
      bu1[v] = wk[v * N + 16];
    }
    v16bf av  = __builtin_bit_cast(v16bf, au);
    v16bf bv0 = __builtin_bit_cast(v16bf, bu0);
    v16bf bv1 = __builtin_bit_cast(v16bf, bu1);
    acc0 = __builtin_amdgcn_wmma_f32_16x16x32_bf16(
        false, av, false, bv0, (short)0, acc0, false, false);
    acc1 = __builtin_amdgcn_wmma_f32_16x16x32_bf16(
        false, av, false, bv1, (short)0, acc1, false, false);
  }
  const int cn0 = tn + l, cn1 = tn + 16 + l;
  const float bb0 = bias ? bias[cn0] : 0.0f;
  const float bb1 = bias ? bias[cn1] : 0.0f;
#pragma unroll
  for (int r = 0; r < 8; ++r) {
    const int cm = tm + r + (half << 3);   // C/D layout: VGPR r -> M=r / M=8+r
    C[cm * ldc + cn0] = acc0[r] + bb0;
    C[cm * ldc + cn1] = acc1[r] + bb1;
  }
}

// ---------------------------------------------------------------------------
// Small WMMA GEMM with on-the-fly f32->bf16 A (per-step M=32 GEMMs only).
// grid = (N/16, M/16), block = 32.
// ---------------------------------------------------------------------------
template <int N, int Kd>
__global__ void gemm_wmma_f32x(const float* __restrict__ X, int ldx,
                               const unsigned* __restrict__ Wpk,
                               const float* __restrict__ bias,
                               float* __restrict__ C, int ldc) {
  const int lane = threadIdx.x & 31;
  const int half = lane >> 4;
  const int l    = lane & 15;
  const int tm   = blockIdx.y << 4;
  const int tn   = blockIdx.x << 4;
  const float* xrow = X + (tm + l) * ldx + (half << 4);
  const unsigned* wcol = Wpk + tn + l + (half << 3) * N;

  v8f acc = {};
#pragma unroll 2
  for (int k0 = 0; k0 < Kd; k0 += 32) {
    v8u au, bu;
    const unsigned* wk = wcol + (k0 >> 1) * N;
#pragma unroll
    for (int v = 0; v < 8; ++v) {
      // A layout: lanes0-15: v0..7->K0..15; lanes16-31 +16.
      const int ka = k0 + 2 * v;
      au[v] = pack2_bf16(xrow[ka], xrow[ka + 1]);
      bu[v] = wk[v * N];
    }
    v16bf av = __builtin_bit_cast(v16bf, au);
    v16bf bv = __builtin_bit_cast(v16bf, bu);
    acc = __builtin_amdgcn_wmma_f32_16x16x32_bf16(
        false, av, false, bv, (short)0, acc, false, false);
  }
  const int cn = tn + l;
  const float bb = bias ? bias[cn] : 0.0f;
#pragma unroll
  for (int r = 0; r < 8; ++r) {
    const int cm = tm + r + (half << 3);
    C[cm * ldc + cn] = acc[r] + bb;
  }
}

// ---------------------------------------------------------------------------
// ypad (U,B,V): step 0 -> one-hot(SOS), step u -> y[:, u-1, :]
// ---------------------------------------------------------------------------
__global__ void ypad_kernel(const float* __restrict__ y, float* __restrict__ ypad) {
  int idx = blockIdx.x * blockDim.x + threadIdx.x;   // U*B*V
  if (idx >= U_ * B_ * V_) return;
  int v = idx & (V_ - 1);
  int b = (idx >> 7) & (B_ - 1);
  int u = idx >> 12;
  ypad[idx] = (u == 0) ? ((v == 0) ? 1.0f : 0.0f)
                       : y[(b * U_ + (u - 1)) * V_ + v];
}

// ---------------------------------------------------------------------------
// init: broadcast h0 to per-batch state, zero logp output
// ---------------------------------------------------------------------------
__global__ void init_kernel(const float* __restrict__ h0in,
                            float* __restrict__ h0buf, float* __restrict__ h1buf,
                            float* __restrict__ logp) {
  int i = blockIdx.x * blockDim.x + threadIdx.x;   // B*D
  if (i >= B_ * D_) return;
  int d = i & (D_ - 1);
  h0buf[i] = h0in[d];
  h1buf[i] = h0in[D_ + d];
  if (i < B_) logp[i] = 0.0f;
}

// ---------------------------------------------------------------------------
// scores + softmax: one block (256 threads) per batch, 8 t's per thread.
// ---------------------------------------------------------------------------
__global__ void attn_kernel(const float* __restrict__ keys,
                            const float* __restrict__ q,
                            float* __restrict__ attn) {
  const int b = blockIdx.x, tid = threadIdx.x;
  __shared__ float sq[K_];
  __shared__ float red[256];
  if (tid < K_) sq[tid] = q[b * K_ + tid];
  __syncthreads();

  float sc[8];
  float mx = -1e30f;
#pragma unroll
  for (int i = 0; i < 8; ++i) {
    const int t = i * 256 + tid;
    const float* kr = keys + (b * T_ + t) * K_;
    float s = 0.0f;
#pragma unroll 4
    for (int k = 0; k < K_; k += 4)
      s += kr[k] * sq[k] + kr[k + 1] * sq[k + 1] +
           kr[k + 2] * sq[k + 2] + kr[k + 3] * sq[k + 3];
    sc[i] = s * 0.08838834764831845f;   // 1/sqrt(128)
    mx = fmaxf(mx, sc[i]);
  }
  red[tid] = mx; __syncthreads();
  for (int s = 128; s > 0; s >>= 1) {
    if (tid < s) red[tid] = fmaxf(red[tid], red[tid + s]);
    __syncthreads();
  }
  mx = red[0]; __syncthreads();

  float sum = 0.0f;
#pragma unroll
  for (int i = 0; i < 8; ++i) { sc[i] = __expf(sc[i] - mx); sum += sc[i]; }
  red[tid] = sum; __syncthreads();
  for (int s = 128; s > 0; s >>= 1) {
    if (tid < s) red[tid] += red[tid + s];
    __syncthreads();
  }
  const float inv = 1.0f / red[0];
#pragma unroll
  for (int i = 0; i < 8; ++i)
    attn[b * T_ + i * 256 + tid] = sc[i] * inv;
}

// ---------------------------------------------------------------------------
// context[b,:] = attn[b,:] @ values[b]; also assembles x = [emb_u, context]
// ---------------------------------------------------------------------------
__global__ void ctx_kernel(const float* __restrict__ attn,
                           const float* __restrict__ values,
                           const float* __restrict__ emb_u,
                           float* __restrict__ xbuf) {
  const int b = blockIdx.x, v = threadIdx.x;   // 256 threads
  const float* a  = attn + b * T_;
  const float* vb = values + (size_t)b * T_ * VAL_;
  float c = 0.0f;
  for (int t = 0; t < T_; t += 4) {
    const float a0 = a[t], a1 = a[t + 1], a2 = a[t + 2], a3 = a[t + 3];
    c += a0 * vb[t * VAL_ + v] +
         a1 * vb[(t + 1) * VAL_ + v] +
         a2 * vb[(t + 2) * VAL_ + v] +
         a3 * vb[(t + 3) * VAL_ + v];
  }
  xbuf[b * (D_ + VAL_) + D_ + v] = c;
  xbuf[b * (D_ + VAL_) + v]      = emb_u[b * D_ + v];
}

// ---------------------------------------------------------------------------
// GRU gate combine (elementwise, in-place h update)
// ---------------------------------------------------------------------------
__global__ void gru_combine_kernel(const float* __restrict__ gi,
                                   const float* __restrict__ gh,
                                   float* __restrict__ h) {
  const int b = blockIdx.x, d = threadIdx.x;   // 256 threads
  const float ir  = gi[b * 768 + d];
  const float iz  = gi[b * 768 + 256 + d];
  const float inn = gi[b * 768 + 512 + d];
  const float hr  = gh[b * 768 + d];
  const float hz  = gh[b * 768 + 256 + d];
  const float hn  = gh[b * 768 + 512 + d];
  const float r = sigmoidf_(ir + hr);
  const float z = sigmoidf_(iz + hz);
  const float n = tanhf(inn + r * hn);
  const float hp = h[b * D_ + d];
  h[b * D_ + d] = (1.0f - z) * n + z * hp;
}

// ---------------------------------------------------------------------------
// log_softmax over V + logp accumulation; one block (V threads) per batch
// ---------------------------------------------------------------------------
__global__ void out_kernel(const float* __restrict__ logits,
                           const float* __restrict__ y, int u,
                           float* __restrict__ logp) {
  const int b = blockIdx.x, v = threadIdx.x;   // 128 threads
  __shared__ float red[V_];
  const float x = logits[b * V_ + v];
  red[v] = x; __syncthreads();
  for (int s = 64; s > 0; s >>= 1) {
    if (v < s) red[v] = fmaxf(red[v], red[v + s]);
    __syncthreads();
  }
  const float mx = red[0]; __syncthreads();
  red[v] = __expf(x - mx); __syncthreads();
  for (int s = 64; s > 0; s >>= 1) {
    if (v < s) red[v] += red[v + s];
    __syncthreads();
  }
  const float lse = __logf(red[0]); __syncthreads();
  const float out = x - mx - lse;
  red[v] = out * y[(b * U_ + u) * V_ + v]; __syncthreads();
  for (int s = 64; s > 0; s >>= 1) {
    if (v < s) red[v] += red[v + s];
    __syncthreads();
  }
  if (v == 0) logp[b] += red[0];
}

// ---------------------------------------------------------------------------
extern "C" void kernel_launch(void* const* d_in, const int* in_sizes, int n_in,
                              void* d_out, int out_size, void* d_ws, size_t ws_size,
                              hipStream_t stream) {
  const float* enc  = (const float*)d_in[0];
  const float* y    = (const float*)d_in[1];
  const float* Wk   = (const float*)d_in[2];
  const float* bk   = (const float*)d_in[3];
  const float* Wq   = (const float*)d_in[4];
  const float* bq   = (const float*)d_in[5];
  const float* Wv   = (const float*)d_in[6];
  const float* bv   = (const float*)d_in[7];
  const float* We   = (const float*)d_in[8];
  const float* be   = (const float*)d_in[9];
  const float* Wih0 = (const float*)d_in[10];
  const float* Whh0 = (const float*)d_in[11];
  const float* bih0 = (const float*)d_in[12];
  const float* bhh0 = (const float*)d_in[13];
  const float* Wih1 = (const float*)d_in[14];
  const float* Whh1 = (const float*)d_in[15];
  const float* bih1 = (const float*)d_in[16];
  const float* bhh1 = (const float*)d_in[17];
  const float* Wout = (const float*)d_in[18];
  const float* bout = (const float*)d_in[19];
  const float* h0in = (const float*)d_in[20];
  float* logp = (float*)d_out;

  // bump allocator over workspace
  char* ws = (char*)d_ws;
  size_t off = 0;
  auto alloc = [&](size_t bytes) -> char* {
    char* p = ws + off;
    off += (bytes + 255) & ~(size_t)255;
    return p;
  };

  float*    keys     = (float*)   alloc((size_t)B_ * T_ * K_ * 4);
  float*    values   = (float*)   alloc((size_t)B_ * T_ * VAL_ * 4);
  unsigned* enc_pk   = (unsigned*)alloc((size_t)B_ * T_ * (ENC_ / 2) * 4);
  unsigned* ypad_pk  = (unsigned*)alloc((size_t)U_ * B_ * (V_ / 2) * 4);
  unsigned* Wk_pk    = (unsigned*)alloc((size_t)(ENC_/2) * K_   * 4);
  unsigned* Wq_pk    = (unsigned*)alloc((size_t)(D_/2)   * K_   * 4);
  unsigned* Wv_pk    = (unsigned*)alloc((size_t)(ENC_/2) * VAL_ * 4);
  unsigned* We_pk    = (unsigned*)alloc((size_t)(V_/2)   * D_   * 4);
  unsigned* Wih0_pk  = (unsigned*)alloc((size_t)(512/2)  * 768  * 4);
  unsigned* Whh0_pk  = (unsigned*)alloc((size_t)(256/2)  * 768  * 4);
  unsigned* Wih1_pk  = (unsigned*)alloc((size_t)(256/2)  * 768  * 4);
  unsigned* Whh1_pk  = (unsigned*)alloc((size_t)(256/2)  * 768  * 4);
  unsigned* Wout_pk  = (unsigned*)alloc((size_t)(D_/2)   * V_   * 4);
  float*    ypad     = (float*)   alloc((size_t)U_ * B_ * V_ * 4);
  float*    emb_all  = (float*)   alloc((size_t)U_ * B_ * D_ * 4);
  float*    qbuf     = (float*)   alloc((size_t)B_ * K_ * 4);
  float*    attn     = (float*)   alloc((size_t)B_ * T_ * 4);
  float*    xbuf     = (float*)   alloc((size_t)B_ * (D_ + VAL_) * 4);
  float*    h0buf    = (float*)   alloc((size_t)B_ * D_ * 4);
  float*    h1buf    = (float*)   alloc((size_t)B_ * D_ * 4);
  float*    gibuf    = (float*)   alloc((size_t)B_ * 768 * 4);
  float*    ghbuf    = (float*)   alloc((size_t)B_ * 768 * 4);

  auto packN = [&](const float* src, unsigned* dst, int N, int Kd) {
    int total = (Kd / 2) * N;
    pack_w_kernel<<<(total + 255) / 256, 256, 0, stream>>>(src, dst, N, Kd);
  };
  // Transposed+bf16-packed weights (Wx is stored (out, in) row-major)
  packN(Wk,   Wk_pk,   K_,   ENC_);
  packN(Wq,   Wq_pk,   K_,   D_);
  packN(Wv,   Wv_pk,   VAL_, ENC_);
  packN(We,   We_pk,   D_,   V_);
  packN(Wih0, Wih0_pk, 768,  512);
  packN(Whh0, Whh0_pk, 768,  256);
  packN(Wih1, Wih1_pk, 768,  256);
  packN(Whh1, Whh1_pk, 768,  256);
  packN(Wout, Wout_pk, V_,   D_);

  {
    int total = U_ * B_ * V_;
    ypad_kernel<<<(total + 255) / 256, 256, 0, stream>>>(y, ypad);
  }
  init_kernel<<<(B_ * D_ + 255) / 256, 256, 0, stream>>>(h0in, h0buf, h1buf, logp);

  // ---- one-shot bf16 packing of big activations ----
  {
    long tp = (long)B_ * T_ * (ENC_ / 2);
    pack_x_kernel<<<(unsigned)((tp + 255) / 256), 256, 0, stream>>>(enc, enc_pk, tp);
    long tp2 = (long)U_ * B_ * (V_ / 2);
    pack_x_kernel<<<(unsigned)((tp2 + 255) / 256), 256, 0, stream>>>(ypad, ypad_pk, tp2);
  }

  // ---- big precompute GEMMs (pre-packed A, 16x32 C per wave) ----
  // keys = enc @ Wk^T + bk          (65536 x 512) x (512 x 128)
  gemm_wmma_pk<K_, ENC_><<<dim3(K_ / 32, (B_ * T_) / 16), 32, 0, stream>>>(
      enc_pk, ENC_ / 2, Wk_pk, bk, keys, K_);
  // values = enc @ Wv^T + bv        (65536 x 512) x (512 x 256)
  gemm_wmma_pk<VAL_, ENC_><<<dim3(VAL_ / 32, (B_ * T_) / 16), 32, 0, stream>>>(
      enc_pk, ENC_ / 2, Wv_pk, bv, values, VAL_);
  // emb_all = ypad @ We^T + be      (2048 x 128) x (128 x 256)
  gemm_wmma_pk<D_, V_><<<dim3(D_ / 32, (U_ * B_) / 16), 32, 0, stream>>>(
      ypad_pk, V_ / 2, We_pk, be, emb_all, D_);

  // ---- sequential decode over U steps ----
  for (int u = 0; u < U_; ++u) {
    // q = h1 @ Wq^T + bq            (32 x 256) x (256 x 128)
    gemm_wmma_f32x<K_, D_><<<dim3(K_ / 16, B_ / 16), 32, 0, stream>>>(
        h1buf, D_, Wq_pk, bq, qbuf, K_);
    // scores + softmax
    attn_kernel<<<B_, 256, 0, stream>>>(keys, qbuf, attn);
    // context + x assembly
    ctx_kernel<<<B_, VAL_, 0, stream>>>(attn, values,
                                        emb_all + (size_t)u * B_ * D_, xbuf);
    // GRU layer 0
    gemm_wmma_f32x<768, D_ + VAL_><<<dim3(768 / 16, B_ / 16), 32, 0, stream>>>(
        xbuf, D_ + VAL_, Wih0_pk, bih0, gibuf, 768);
    gemm_wmma_f32x<768, D_><<<dim3(768 / 16, B_ / 16), 32, 0, stream>>>(
        h0buf, D_, Whh0_pk, bhh0, ghbuf, 768);
    gru_combine_kernel<<<B_, D_, 0, stream>>>(gibuf, ghbuf, h0buf);
    // GRU layer 1
    gemm_wmma_f32x<768, D_><<<dim3(768 / 16, B_ / 16), 32, 0, stream>>>(
        h0buf, D_, Wih1_pk, bih1, gibuf, 768);
    gemm_wmma_f32x<768, D_><<<dim3(768 / 16, B_ / 16), 32, 0, stream>>>(
        h1buf, D_, Whh1_pk, bhh1, ghbuf, 768);
    gru_combine_kernel<<<B_, D_, 0, stream>>>(gibuf, ghbuf, h1buf);
    // logits + log_softmax + logp accumulation
    gemm_wmma_f32x<V_, D_><<<dim3(V_ / 16, B_ / 16), 32, 0, stream>>>(
        h1buf, D_, Wout_pk, bout, gibuf /*reuse as logits*/, V_);
    out_kernel<<<B_, V_, 0, stream>>>(gibuf, y, u, logp);
  }
}
